// HiGCIN_volleyball_13494787244859
// MI455X (gfx1250) — matrix operand order repaired and checked by reading, not compiled, not run on hardware
//
#include <hip/hip_runtime.h>

typedef __attribute__((ext_vector_type(16))) __bf16 v16bf;
typedef __attribute__((ext_vector_type(8)))  float  v8f;

#define BB 96
#define TT 10
#define NNODE 25
#define DD 1056
#define LLAT 528
#define MROWS (BB*TT*NNODE)     // 24000
#define PSTRIDE (3*LLAT)        // 1584: fused [th|ph|g] row stride

union FragU {
    uint4 q[2];
    v16bf v;
};

// ---------------- fp32 -> bf16 pre-conversion (memory bound) ----------------
__global__ __launch_bounds__(256)
void f32_to_bf16(const float* __restrict__ src, __bf16* __restrict__ dst, long long n) {
    long long i = ((long long)blockIdx.x * 256 + threadIdx.x) * 8;
    if (i + 8 <= n) {
        float4 a = *(const float4*)(src + i);
        float4 b = *(const float4*)(src + i + 4);
        union { __bf16 h[8]; uint4 q; } u;
        u.h[0] = (__bf16)a.x; u.h[1] = (__bf16)a.y; u.h[2] = (__bf16)a.z; u.h[3] = (__bf16)a.w;
        u.h[4] = (__bf16)b.x; u.h[5] = (__bf16)b.y; u.h[6] = (__bf16)b.z; u.h[7] = (__bf16)b.w;
        *(uint4*)(dst + i) = u.q;
    } else {
        for (; i < n; ++i) dst[i] = (__bf16)src[i];
    }
}

// ---------------- GEMM helpers ----------------
__device__ __forceinline__ void prefetch_tiles(const __bf16* __restrict__ A,
                                               const __bf16* __restrict__ W,
                                               int m0, int n0, int K, int Nout,
                                               int k0, int tid, bool fullN,
                                               uint4 pa[3], uint4 pb[4]) {
    const bool fullK = (k0 + 64 <= K);
    #pragma unroll
    for (int t = 0; t < 3; ++t) {                 // A tile 96x64: 768 16B chunks
        int idx = tid + t * 256;
        int r = idx >> 3, c8 = (idx & 7) << 3;
        int kk = k0 + c8;
        pa[t] = make_uint4(0u, 0u, 0u, 0u);
        if (fullK || kk < K) pa[t] = *(const uint4*)(A + (size_t)(m0 + r) * K + kk);
    }
    #pragma unroll
    for (int t = 0; t < 4; ++t) {                 // B tile 128x64: 1024 16B chunks
        int idx = tid + t * 256;
        int r = idx >> 3, c8 = (idx & 7) << 3;
        int kk = k0 + c8;
        pb[t] = make_uint4(0u, 0u, 0u, 0u);
        if ((fullN || (n0 + r) < Nout) && (fullK || kk < K))
            pb[t] = *(const uint4*)(W + (size_t)(n0 + r) * K + kk);
    }
}

__device__ __forceinline__ void store_tiles(__bf16* __restrict__ AsB, __bf16* __restrict__ BsB,
                                            int tid, const uint4 pa[3], const uint4 pb[4]) {
    #pragma unroll
    for (int t = 0; t < 3; ++t) {
        int idx = tid + t * 256;
        int r = idx >> 3, c8 = (idx & 7) << 3;
        *(uint4*)(&AsB[r * 72 + c8]) = pa[t];
    }
    #pragma unroll
    for (int t = 0; t < 4; ++t) {
        int idx = tid + t * 256;
        int r = idx >> 3, c8 = (idx & 7) << 3;
        *(uint4*)(&BsB[r * 72 + c8]) = pb[t];
    }
}

// ---------------- bf16 WMMA GEMM:  C[M,Nout] = A[M,K] x W[Nout,K]^T (+resid) ----------------
// 256 threads = 8 waves, block tile 96(M) x 128(N), K-step 64, double-buffered LDS.
// Wave grid 2(M) x 4(N): wave owns 48x32 => 12 WMMA per K-step, 1 barrier per K-step.
__global__ __launch_bounds__(256)
void gemm_bf16_wmma(const __bf16* __restrict__ A, const __bf16* __restrict__ W,
                    const float* __restrict__ resid, float* __restrict__ C,
                    int M, int Nout, int K) {
    __shared__ __align__(16) __bf16 As[2][96 * 72];    // 2 x 13.5 KB
    __shared__ __align__(16) __bf16 Bs[2][128 * 72];   // 2 x 18 KB

    const int tid   = threadIdx.x;
    const int lane  = tid & 31;
    const int wave  = tid >> 5;
    const int waveM = wave & 1;
    const int waveN = wave >> 1;
    const int m0 = blockIdx.x * 96;
    const int n0 = blockIdx.y * 128;
    const int khalf = lane >> 4;
    const int l15   = lane & 15;
    const bool fullN = (n0 + 128 <= Nout);
    const int nsteps = (K + 63) >> 6;

    v8f acc[3][2];
    #pragma unroll
    for (int i = 0; i < 3; ++i)
        #pragma unroll
        for (int j = 0; j < 2; ++j)
            acc[i][j] = (v8f){0.f,0.f,0.f,0.f,0.f,0.f,0.f,0.f};

    uint4 pa[3], pb[4];
    prefetch_tiles(A, W, m0, n0, K, Nout, 0, tid, fullN, pa, pb);
    store_tiles(As[0], Bs[0], tid, pa, pb);
    __syncthreads();

    for (int s = 0; s < nsteps; ++s) {
        const int buf = s & 1;
        const bool has_next = (s + 1 < nsteps);
        if (has_next)
            prefetch_tiles(A, W, m0, n0, K, Nout, (s + 1) * 64, tid, fullN, pa, pb);

        const __bf16* AsB = As[buf];
        const __bf16* BsB = Bs[buf];
        #pragma unroll
        for (int ks = 0; ks < 2; ++ks) {          // two K=32 sub-steps
            FragU fb[2];
            #pragma unroll
            for (int j = 0; j < 2; ++j) {
                const __bf16* bp = &BsB[(waveN * 32 + j * 16 + l15) * 72 + ks * 32 + khalf * 8];
                fb[j].q[0] = *(const uint4*)(bp);
                fb[j].q[1] = *(const uint4*)(bp + 16);
            }
            #pragma unroll
            for (int i = 0; i < 3; ++i) {
                FragU fa;
                const __bf16* ap = &AsB[(waveM * 48 + i * 16 + l15) * 72 + ks * 32 + khalf * 8];
                fa.q[0] = *(const uint4*)(ap);
                fa.q[1] = *(const uint4*)(ap + 16);
                acc[i][0] = __builtin_amdgcn_wmma_f32_16x16x32_bf16(false, fa.v, false, fb[0].v, (short)0, acc[i][0], false, false);
                acc[i][1] = __builtin_amdgcn_wmma_f32_16x16x32_bf16(false, fa.v, false, fb[1].v, (short)0, acc[i][1], false, false);
            }
        }

        if (has_next) {
            store_tiles(As[buf ^ 1], Bs[buf ^ 1], tid, pa, pb);
            __syncthreads();
        }
    }

    // store: elem r -> M = base + khalf*8 + r, N = l15
    #pragma unroll
    for (int i = 0; i < 3; ++i) {
        const int mrow = m0 + waveM * 48 + i * 16 + khalf * 8;
        #pragma unroll
        for (int j = 0; j < 2; ++j) {
            const int col = n0 + waveN * 32 + j * 16 + l15;
            if (fullN || col < Nout) {
                #pragma unroll
                for (int r = 0; r < 8; ++r) {
                    size_t o = (size_t)(mrow + r) * Nout + col;
                    float v = acc[i][j][r];
                    if (resid) v += resid[o];
                    C[o] = v;
                }
            }
        }
    }
}

// ---------------- spatial cross-inference: one block per (b,t) frame ----------------
__global__ __launch_bounds__(256)
void spatial_kernel(const float* __restrict__ proj, float* __restrict__ spatio) {
    __shared__ float phs[NNODE * LLAT];
    __shared__ float sw[NNODE * NNODE];
    const int bt = blockIdx.x;
    const size_t r0 = (size_t)bt * NNODE;
    const int tid = threadIdx.x;

    for (int i = tid; i < NNODE * LLAT; i += 256) {
        int s = i / LLAT, l = i % LLAT;
        phs[i] = proj[(r0 + s) * PSTRIDE + LLAT + l];
    }
    __syncthreads();

    for (int p = tid; p < NNODE * NNODE; p += 256) {
        int j = p / NNODE, k = p % NNODE;
        const float* tr = proj + (r0 + j) * PSTRIDE;
        const float* pr = &phs[k * LLAT];
        float s = 0.f;
        #pragma unroll 4
        for (int l = 0; l < LLAT; ++l) s += tr[l] * pr[l];
        sw[p] = s * (1.0f / NNODE);
    }
    __syncthreads();

    for (int idx = tid; idx < NNODE * LLAT; idx += 256) {
        int j = idx / LLAT, l = idx % LLAT;
        float s = 0.f;
        #pragma unroll
        for (int k = 0; k < NNODE; ++k)
            s += sw[j * NNODE + k] * proj[(r0 + k) * PSTRIDE + 2 * LLAT + l];
        spatio[(r0 + j) * LLAT + l] = s;
    }
}

// ---------------- temporal cross-inference + combine: one block per (b,j) node ----------------
__global__ __launch_bounds__(256)
void temporal_kernel(const float* __restrict__ proj, const float* __restrict__ spatio,
                     __bf16* __restrict__ stb) {
    __shared__ float phs[TT * LLAT];
    __shared__ float gs[TT * LLAT];
    __shared__ float tw[TT * TT];
    const int bn = blockIdx.x;            // b*25 + j
    const int b = bn / NNODE, j = bn % NNODE;
    const int tid = threadIdx.x;

    for (int i = tid; i < TT * LLAT; i += 256) {
        int s = i / LLAT, l = i % LLAT;
        size_t row = (size_t)(b * TT + s) * NNODE + j;
        phs[i] = proj[row * PSTRIDE + LLAT + l];
        gs[i]  = proj[row * PSTRIDE + 2 * LLAT + l];
    }
    __syncthreads();

    for (int p = tid; p < TT * TT; p += 256) {
        int t = p / TT, s = p % TT;
        size_t rt = (size_t)(b * TT + t) * NNODE + j;
        const float* tr = proj + rt * PSTRIDE;
        const float* pr = &phs[s * LLAT];
        float acc = 0.f;
        #pragma unroll 4
        for (int l = 0; l < LLAT; ++l) acc += tr[l] * pr[l];
        tw[p] = acc * (1.0f / TT);
    }
    __syncthreads();

    const float sc = 1.0f / (float)(TT + NNODE);
    for (int idx = tid; idx < TT * LLAT; idx += 256) {
        int t = idx / LLAT, l = idx % LLAT;
        float acc = 0.f;
        #pragma unroll
        for (int s = 0; s < TT; ++s) acc += tw[t * TT + s] * gs[s * LLAT + l];
        size_t rt = (size_t)(b * TT + t) * NNODE + j;
        stb[rt * LLAT + l] = (__bf16)((spatio[rt * LLAT + l] + acc) * sc);
    }
}

// ---------------- BatchNorm ----------------
__global__ __launch_bounds__(256)
void bn_stats(const float* __restrict__ out, float* __restrict__ stats) {
    __shared__ float ssum[256], ssq[256];
    const int c  = threadIdx.x & 15;
    const int rg = threadIdx.x >> 4;
    const int d  = blockIdx.x * 16 + c;
    float s = 0.f, q = 0.f;
    for (int row = rg; row < MROWS; row += 16) {
        float v = out[(size_t)row * DD + d];
        s += v; q += v * v;
    }
    ssum[threadIdx.x] = s; ssq[threadIdx.x] = q;
    __syncthreads();
    for (int h = 8; h > 0; h >>= 1) {
        if (rg < h) {
            ssum[threadIdx.x] += ssum[threadIdx.x + h * 16];
            ssq[threadIdx.x]  += ssq[threadIdx.x + h * 16];
        }
        __syncthreads();
    }
    if (rg == 0) {
        float mean = ssum[c] * (1.0f / MROWS);
        float var  = ssq[c] * (1.0f / MROWS) - mean * mean;
        stats[d]      = mean;
        stats[DD + d] = rsqrtf(var + 1e-5f);
    }
}

__global__ __launch_bounds__(256)
void bn_apply(float* __restrict__ out, const float* __restrict__ stats,
              const float* __restrict__ gamma, const float* __restrict__ beta) {
    size_t i4 = (size_t)blockIdx.x * 256 + threadIdx.x;
    size_t flat = i4 * 4;
    if (flat >= (size_t)MROWS * DD) return;
    float4 v = *(float4*)(out + flat);
    int d0 = (int)(flat % DD);
    float r[4] = {v.x, v.y, v.z, v.w};
    #pragma unroll
    for (int i = 0; i < 4; ++i) {
        int d = d0 + i;
        r[i] = (r[i] - stats[d]) * stats[DD + d] * gamma[d] + beta[d];
    }
    v.x = r[0]; v.y = r[1]; v.z = r[2]; v.w = r[3];
    *(float4*)(out + flat) = v;
}

// ---------------- launcher ----------------
extern "C" void kernel_launch(void* const* d_in, const int* in_sizes, int n_in,
                              void* d_out, int out_size, void* d_ws, size_t ws_size,
                              hipStream_t stream) {
    const float* x      = (const float*)d_in[0];
    const float* Wtheta = (const float*)d_in[1];
    const float* Wphi   = (const float*)d_in[2];
    const float* Wg     = (const float*)d_in[3];
    const float* Wout   = (const float*)d_in[4];
    const float* gamma  = (const float*)d_in[5];
    const float* beta   = (const float*)d_in[6];
    float* out = (float*)d_out;
    float* ws  = (float*)d_ws;

    const size_t seg = (size_t)MROWS * LLAT;          // 12,672,000 floats
    float*  proj   = ws;                              // [24000 x 1584] fp32
    float*  spatio = ws + 3 * seg;                    // [24000 x 528]  fp32
    __bf16* xb     = (__bf16*)(ws + 4 * seg);         // [24000 x 1056] bf16
    __bf16* stb    = (__bf16*)(ws + 4 * seg);         // overlaps xb (dead by then)
    float*  wtail  = ws + 5 * seg;
    __bf16* wallb  = (__bf16*)wtail;                  // [1584 x 1056] bf16
    __bf16* woutb  = (__bf16*)(wtail + ((size_t)PSTRIDE * DD + 1) / 2);
    float*  stats  = wtail + ((size_t)PSTRIDE * DD + 1) / 2 + ((size_t)DD * LLAT + 1) / 2;

    const long long nx = (long long)MROWS * DD;       // 25,344,000
    const long long nw = (long long)LLAT * DD;        // 557,568

    f32_to_bf16<<<(int)((nx / 8 + 255) / 256), 256, 0, stream>>>(x, xb, nx);
    f32_to_bf16<<<(int)((nw / 8 + 255) / 256), 256, 0, stream>>>(Wtheta, wallb, nw);
    f32_to_bf16<<<(int)((nw / 8 + 255) / 256), 256, 0, stream>>>(Wphi,   wallb + nw, nw);
    f32_to_bf16<<<(int)((nw / 8 + 255) / 256), 256, 0, stream>>>(Wg,     wallb + 2 * nw, nw);
    f32_to_bf16<<<(int)((nw / 8 + 255) / 256), 256, 0, stream>>>(Wout,   woutb, nw);

    dim3 blk(256);
    dim3 gp(MROWS / 96, (PSTRIDE + 127) / 128);       // 250 x 13
    gemm_bf16_wmma<<<gp, blk, 0, stream>>>(xb, wallb, nullptr, proj, MROWS, PSTRIDE, DD);

    spatial_kernel<<<BB * TT, 256, 0, stream>>>(proj, spatio);
    temporal_kernel<<<BB * NNODE, 256, 0, stream>>>(proj, spatio, stb);

    dim3 go(MROWS / 96, (DD + 127) / 128);            // 250 x 9
    gemm_bf16_wmma<<<go, blk, 0, stream>>>(stb, woutb, x, out, MROWS, DD, LLAT);

    bn_stats<<<DD / 16, 256, 0, stream>>>(out, stats);
    bn_apply<<<(int)(((size_t)MROWS * DD / 4 + 255) / 256), 256, 0, stream>>>(out, stats, gamma, beta);
}
